// IsgnOldGraphSingleEncoder_10033043603833
// MI455X (gfx1250) — matrix operand order repaired and checked by reading, not compiled, not run on hardware
//
#include <hip/hip_runtime.h>
#include <math.h>

// ---------------- static config (mirrors reference) ----------------
#define N_NOTES 1024
#define IN_SIZE 78
#define IN_PAD  96             // padded K for note_fc (multiple of 32)
#define NOTE_H  256
#define MEAS_H  256
#define GSIZE   768
#define N_EDGE  8
#define N_HEAD  8
#define ATT_D   512
#define NPM     16
#define N_MEAS  64
#define HS      (N_NOTES * GSIZE)        // 786432
#define KAGG    (N_EDGE * GSIZE)         // 6144

typedef __attribute__((ext_vector_type(16))) _Float16 v16h;
typedef __attribute__((ext_vector_type(8)))  _Float16 h8;
typedef __attribute__((ext_vector_type(8)))  float    v8f;

__device__ __forceinline__ float sigf(float x) { return 1.0f / (1.0f + __expf(-x)); }

__device__ __forceinline__ v16h ldfrag(const _Float16* p, int off2)
{
    h8 lo = *(const h8*)(p);
    h8 hi = *(const h8*)(p + off2);
    return __builtin_shufflevector(lo, hi, 0,1,2,3,4,5,6,7,8,9,10,11,12,13,14,15);
}

// =====================================================================
// WMMA GEMM, all-f16 operands, k-contiguous, no bounds checks.
//   C_e[m,n] (+)= sum_k A_e[m][k] * B_e[n][k]   (B stored transposed: [N][K])
// Requirements: M,N multiples of 32; K multiple of 32; 16B-aligned rows.
// One wave = one 32x32 C block (2x2 WMMA tiles): per 32-K step it loads
// 2 A-frags + 2 B-frags (8 x b128) and issues 4 independent WMMAs
// (2x register-level operand reuse vs 1 tile/wave).
// =====================================================================
__global__ __launch_bounds__(128)
void wmma_gemm16(const _Float16* __restrict__ A, long lda, long bA,
                 const _Float16* __restrict__ B, long ldb, long bB,
                 float* __restrict__ C, long ldc, long bC,
                 _Float16* __restrict__ C16, long ldc16, long bC16,
                 const float* __restrict__ bias, long bBias,
                 int M, int N, int K, int accum, int act)
{
    const int e  = blockIdx.z;
    const int tn = blockIdx.x * 4 + threadIdx.y;   // 32-col block index (uniform per wave)
    const int tm = blockIdx.y;                     // 32-row block index
    if (tn * 32 >= N) return;                      // whole wave exits together

    const int lane = threadIdx.x;                  // wave32
    const int hi   = lane >> 4;                    // half-wave select
    const int l16  = lane & 15;

    // A frags: rows tm*32+l16 and +16; halves [ka..ka+7],[ka+16..ka+23], ka = k0 + hi*8
    const _Float16* ap0 = A + (long)e * bA + (long)(tm * 32 + l16) * lda + hi * 8;
    const _Float16* ap1 = ap0 + 16 * lda;
    // B frags: cols tn*32+l16 and +16; halves [kb..kb+7],[kb+16..kb+23], kb = k0 + hi*8
    const _Float16* bp0 = B + (long)e * bB + (long)(tn * 32 + l16) * ldb + hi * 8;
    const _Float16* bp1 = bp0 + 16 * ldb;

    v8f acc00 = {0.f,0.f,0.f,0.f,0.f,0.f,0.f,0.f};
    v8f acc01 = acc00, acc10 = acc00, acc11 = acc00;
    if (accum) {
        const float* Ce = C + (long)e * bC;
#pragma unroll
        for (int v = 0; v < 8; ++v) {
            long r0 = (long)(tm * 32 + v + hi * 8) * ldc;
            long r1 = r0 + 16 * ldc;
            int  c0 = tn * 32 + l16;
            acc00[v] = Ce[r0 + c0];      acc01[v] = Ce[r0 + c0 + 16];
            acc10[v] = Ce[r1 + c0];      acc11[v] = Ce[r1 + c0 + 16];
        }
    }

#pragma unroll 2
    for (int k0 = 0; k0 < K; k0 += 32) {
        v16h a0 = ldfrag(ap0 + k0, 16);
        v16h a1 = ldfrag(ap1 + k0, 16);
        v16h b0 = ldfrag(bp0 + k0, 16);
        v16h b1 = ldfrag(bp1 + k0, 16);
        acc00 = __builtin_amdgcn_wmma_f32_16x16x32_f16(false, a0, false, b0, (short)0, acc00, false, false);
        acc01 = __builtin_amdgcn_wmma_f32_16x16x32_f16(false, a0, false, b1, (short)0, acc01, false, false);
        acc10 = __builtin_amdgcn_wmma_f32_16x16x32_f16(false, a1, false, b0, (short)0, acc10, false, false);
        acc11 = __builtin_amdgcn_wmma_f32_16x16x32_f16(false, a1, false, b1, (short)0, acc11, false, false);
    }

    const int col0 = tn * 32 + l16, col1 = col0 + 16;
    float bv0 = bias ? bias[(long)e * bBias + col0] : 0.f;
    float bv1 = bias ? bias[(long)e * bBias + col1] : 0.f;
#pragma unroll
    for (int v = 0; v < 8; ++v) {
        int row0 = tm * 32 + v + hi * 8;
        int row1 = row0 + 16;
        float r00 = acc00[v] + bv0, r01 = acc01[v] + bv1;
        float r10 = acc10[v] + bv0, r11 = acc11[v] + bv1;
        if (act == 1) {
            r00 = fmaxf(r00, 0.f); r01 = fmaxf(r01, 0.f);
            r10 = fmaxf(r10, 0.f); r11 = fmaxf(r11, 0.f);
        } else if (act == 2) {
            r00 = tanhf(r00); r01 = tanhf(r01);
            r10 = tanhf(r10); r11 = tanhf(r11);
        }
        if (C) {
            float* Ce = C + (long)e * bC;
            Ce[(long)row0 * ldc + col0] = r00;  Ce[(long)row0 * ldc + col1] = r01;
            Ce[(long)row1 * ldc + col0] = r10;  Ce[(long)row1 * ldc + col1] = r11;
        }
        if (C16) {
            _Float16* Cs = C16 + (long)e * bC16;
            Cs[(long)row0 * ldc16 + col0] = (_Float16)r00;  Cs[(long)row0 * ldc16 + col1] = (_Float16)r01;
            Cs[(long)row1 * ldc16 + col0] = (_Float16)r10;  Cs[(long)row1 * ldc16 + col1] = (_Float16)r11;
        }
    }
}

// ---------------- layout transforms (cheap bandwidth passes) ----------------
// f32 [b][R][C] -> f16 transposed: out[boff + c*outLd + r], boff=(b/grp)*sHi+(b%grp)*sLo
__global__ __launch_bounds__(256)
void transpose_f32f16_k(const float* __restrict__ in, long inBatch, int R, int C,
                        _Float16* __restrict__ out, long outLd,
                        int grp, long sHi, long sLo)
{
    __shared__ float tile[32][33];
    const int b = blockIdx.z;
    const float* ib = in + (long)b * inBatch;
    _Float16* ob = out + (long)(b / grp) * sHi + (long)(b % grp) * sLo;
    const int r0 = blockIdx.y * 32, c0 = blockIdx.x * 32;
#pragma unroll
    for (int i = threadIdx.y; i < 32; i += 8) {
        int r = r0 + i, c = c0 + threadIdx.x;
        tile[i][threadIdx.x] = (r < R && c < C) ? ib[(long)r * C + c] : 0.f;
    }
    __syncthreads();
#pragma unroll
    for (int i = threadIdx.y; i < 32; i += 8) {
        int c = c0 + i, r = r0 + threadIdx.x;
        if (c < C && r < R) ob[(long)c * outLd + r] = (_Float16)tile[threadIdx.x][i];
    }
}

// f32 [R][C] -> f16 [R][Cp] (zero pad cols C..Cp-1)
__global__ void convert_pad_k(const float* __restrict__ in, int R, int C,
                              _Float16* __restrict__ out, int Cp)
{
    int i = blockIdx.x * blockDim.x + threadIdx.x;
    if (i >= R * Cp) return;
    int r = i / Cp, c = i % Cp;
    out[i] = (c < C) ? (_Float16)in[r * C + c] : (_Float16)0.f;
}

__global__ void fill_zero_k(float* p, int n)
{
    int i = blockIdx.x * blockDim.x + threadIdx.x;
    if (i < n) p[i] = 0.f;
}

__global__ void copy_k(const float* __restrict__ s, float* __restrict__ d, int n)
{
    int i = blockIdx.x * blockDim.x + threadIdx.x;
    if (i < n) d[i] = s[i];
}

// ---------------- GRU gate elementwise ----------------
__global__ void gate_zr_k(float* __restrict__ agg, const float* __restrict__ gb,
                          const float* __restrict__ h, _Float16* __restrict__ rh16)
{
    int i = blockIdx.x * blockDim.x + threadIdx.x;
    if (i >= HS) return;
    int col = i % GSIZE;
    float z = sigf(agg[i] + gb[col]);
    float r = sigf(agg[HS + i] + gb[GSIZE + col]);
    agg[i]  = z;
    rh16[i] = (_Float16)(r * h[i]);
}

__global__ void gate_update_k(const float* __restrict__ agg, const float* __restrict__ gb,
                              float* __restrict__ h)
{
    int i = blockIdx.x * blockDim.x + threadIdx.x;
    if (i >= HS) return;
    int col = i % GSIZE;
    float ht = tanhf(agg[2 * HS + i] + gb[2 * GSIZE + col]);
    float z  = agg[i];
    h[i] = z * h[i] + (1.f - z) * ht;
}

// cat[n,0:256] = h1[n,512:768]; cat[n,256:512] = h2[n,512:768]  (f32 + f16 copies)
__global__ void build_cat_k(const float* __restrict__ h1, const float* __restrict__ h2,
                            float* __restrict__ cat, _Float16* __restrict__ cat16)
{
    int i = blockIdx.x * blockDim.x + threadIdx.x;
    if (i >= N_NOTES * ATT_D) return;
    int n = i >> 9, c = i & 511;
    float v = (c < 256) ? h1[n * GSIZE + 512 + c] : h2[n * GSIZE + 256 + c];
    cat[i] = v;
    cat16[i] = (_Float16)v;
}

// s[n,h] = sum_d tanh_a[n, h*64+d] * cv[h,d]
__global__ void att_sim_k(const float* __restrict__ a, const float* __restrict__ cv,
                          float* __restrict__ s)
{
    int i = blockIdx.x * blockDim.x + threadIdx.x;
    if (i >= N_NOTES * N_HEAD) return;
    int n = i >> 3, h = i & 7;
    float acc = 0.f;
#pragma unroll 8
    for (int d = 0; d < 64; ++d) acc += a[n * ATT_D + h * 64 + d] * cv[h * 64 + d];
    s[i] = acc;
}

// per-measure softmax over 16 notes, per head; pooled = weighted sum of cat
__global__ __launch_bounds__(256)
void att_pool_k(const float* __restrict__ s, const float* __restrict__ cat,
                float* __restrict__ pooled)
{
    int m = blockIdx.x;
    __shared__ float w[NPM][N_HEAD];
    __shared__ float inv[N_HEAD];
    int tid = threadIdx.x;
    if (tid < NPM * N_HEAD) {
        int i = tid >> 3, h = tid & 7;
        w[i][h] = s[(m * NPM + i) * N_HEAD + h];
    }
    __syncthreads();
    if (tid < N_HEAD) {
        float mx = -1e30f;
        for (int i = 0; i < NPM; ++i) mx = fmaxf(mx, w[i][tid]);
        float sum = 0.f;
        for (int i = 0; i < NPM; ++i) { float e = __expf(w[i][tid] - mx); w[i][tid] = e; sum += e; }
        inv[tid] = 1.f / sum;
    }
    __syncthreads();
    if (tid < NPM * N_HEAD) { int i = tid >> 3, h = tid & 7; w[i][h] *= inv[h]; }
    __syncthreads();
    for (int d = tid; d < ATT_D; d += 256) {
        int h = d >> 6;
        float acc = 0.f;
#pragma unroll
        for (int i = 0; i < NPM; ++i) acc += w[i][h] * cat[(m * NPM + i) * ATT_D + d];
        pooled[m * ATT_D + d] = acc;
    }
}

// one block per direction; 64-step LSTM; xih pre-holds x@Wih^T + bih
__global__ __launch_bounds__(256)
void lstm_scan_k(const float* __restrict__ xih, const float* __restrict__ Whh,
                 const float* __restrict__ bhh, float* __restrict__ out)
{
    const int d = blockIdx.x;
    const float* xd = xih + (long)d * N_MEAS * 1024;
    const float* W  = Whh + (long)d * 1024 * MEAS_H;
    const float* bb = bhh + (long)d * 1024;
    __shared__ float hs[MEAS_H];
    __shared__ float cs[MEAS_H];
    const int tid = threadIdx.x;
    hs[tid] = 0.f; cs[tid] = 0.f;
    __syncthreads();
    for (int t = 0; t < N_MEAS; ++t) {
        const int tt = d ? (N_MEAS - 1 - t) : t;
        float gi = xd[tt * 1024 + tid]       + bb[tid];
        float gf = xd[tt * 1024 + 256 + tid] + bb[256 + tid];
        float gg = xd[tt * 1024 + 512 + tid] + bb[512 + tid];
        float go = xd[tt * 1024 + 768 + tid] + bb[768 + tid];
        const float* w0 = W + (long)tid * MEAS_H;
        const float* w1 = W + (long)(256 + tid) * MEAS_H;
        const float* w2 = W + (long)(512 + tid) * MEAS_H;
        const float* w3 = W + (long)(768 + tid) * MEAS_H;
        for (int k = 0; k < MEAS_H; ++k) {
            float hk = hs[k];
            gi += w0[k] * hk; gf += w1[k] * hk; gg += w2[k] * hk; go += w3[k] * hk;
        }
        float c = sigf(gf) * cs[tid] + sigf(gi) * tanhf(gg);
        float h = sigf(go) * tanhf(c);
        __syncthreads();
        hs[tid] = h; cs[tid] = c;
        out[tt * (2 * MEAS_H) + d * MEAS_H + tid] = h;
        __syncthreads();
    }
}

// final_out[n,:1536] = [mh[seg(n),0:512], h1[n,512:768], h2[n,0:768]]; then mh flat
__global__ void write_out_k(const float* __restrict__ h1, const float* __restrict__ h2,
                            const float* __restrict__ mh, const int* __restrict__ mnum,
                            float* __restrict__ out)
{
    const int TOT1 = N_NOTES * 1536;
    const int TOT  = TOT1 + N_MEAS * 512;
    int i = blockIdx.x * blockDim.x + threadIdx.x;
    if (i >= TOT) return;
    if (i < TOT1) {
        int n = i / 1536, c = i % 1536;
        int m = mnum[n] - mnum[0];
        float v;
        if (c < 512)      v = mh[m * 512 + c];
        else if (c < 768) v = h1[n * GSIZE + 512 + (c - 512)];
        else              v = h2[n * GSIZE + (c - 768)];
        out[i] = v;
    } else {
        out[i] = mh[i - TOT1];
    }
}

// ---------------- host helpers ----------------
static void gemm16(hipStream_t st,
                   const _Float16* A, long lda, long bA,
                   const _Float16* B, long ldb, long bB,
                   float* C, long ldc, long bC,
                   _Float16* C16, long ldc16, long bC16,
                   const float* bias, long bBias,
                   int M, int N, int K, int accum, int act, int batch)
{
    dim3 grid(((N + 31) / 32 + 3) / 4, (M + 31) / 32, batch), block(32, 4, 1);
    wmma_gemm16<<<grid, block, 0, st>>>(A, lda, bA, B, ldb, bB, C, ldc, bC,
                                        C16, ldc16, bC16, bias, bBias, M, N, K, accum, act);
}

static void transpose16(hipStream_t st, const float* in, long inBatch, int R, int C,
                        _Float16* out, long outLd, int grp, long sHi, long sLo, int batches)
{
    dim3 grid((C + 31) / 32, (R + 31) / 32, batches), block(32, 8, 1);
    transpose_f32f16_k<<<grid, block, 0, st>>>(in, inBatch, R, C, out, outLd, grp, sHi, sLo);
}

static void convert16(hipStream_t st, const float* in, int R, int C, _Float16* out, int Cp)
{
    int tot = R * Cp;
    convert_pad_k<<<(tot + 255) / 256, 256, 0, st>>>(in, R, C, out, Cp);
}

struct GraphBufs {
    const _Float16* adjT16;
    _Float16 *wT16, *uT16, *act16, *h16, *hT16, *rh16;
    float *agg;
};

static void graph_pass(hipStream_t st, const float* gw, const float* gu, const float* gb,
                       float* h, const GraphBufs& G)
{
    const long SS = (long)GSIZE * GSIZE;
    // weight transposes for this graph (once per pass)
    transpose16(st, gw, SS, GSIZE, GSIZE, G.wT16, KAGG, N_EDGE,
                (long)GSIZE * KAGG, GSIZE, 3 * N_EDGE);               // wT[g][k][e*768+s]
    transpose16(st, gu, SS, GSIZE, GSIZE, G.uT16, GSIZE, 1, SS, 0, 3); // uT[g][k][s]
    const int EB = (HS + 255) / 256;
    for (int it = 0; it < 2; ++it) {
        convert16(st, h, N_NOTES, GSIZE, G.h16, GSIZE);                     // h16[n][s]
        transpose16(st, h, 0, N_NOTES, GSIZE, G.hT16, N_NOTES, 1, 0, 0, 1); // hT16[s][n]
        // act16[n][e*768+s] = sum_m adjT[e][n][m] * hT[s][m]
        gemm16(st, G.adjT16, N_NOTES, (long)N_NOTES * N_NOTES,
               G.hT16, N_NOTES, 0,
               nullptr, 0, 0, G.act16, KAGG, GSIZE, nullptr, 0,
               N_NOTES, GSIZE, N_NOTES, 0, 0, N_EDGE);
        // agg[g] = act16 @ wT[g]^T   (K = 6144)
        gemm16(st, G.act16, KAGG, 0,
               G.wT16, KAGG, (long)GSIZE * KAGG,
               G.agg, GSIZE, HS, nullptr, 0, 0, nullptr, 0,
               N_NOTES, GSIZE, KAGG, 0, 0, 3);
        // agg[0] += h@u0 ; agg[1] += h@u1
        gemm16(st, G.h16, GSIZE, 0,
               G.uT16, GSIZE, SS,
               G.agg, GSIZE, HS, nullptr, 0, 0, nullptr, 0,
               N_NOTES, GSIZE, GSIZE, 1, 0, 2);
        gate_zr_k<<<EB, 256, 0, st>>>(G.agg, gb, h, G.rh16);
        // agg[2] += (r*h) @ u2
        gemm16(st, G.rh16, GSIZE, 0,
               G.uT16 + 2 * SS, GSIZE, 0,
               G.agg + 2 * (long)HS, GSIZE, 0, nullptr, 0, 0, nullptr, 0,
               N_NOTES, GSIZE, GSIZE, 1, 0, 1);
        gate_update_k<<<EB, 256, 0, st>>>(G.agg, gb, h);
    }
}

extern "C" void kernel_launch(void* const* d_in, const int* in_sizes, int n_in,
                              void* d_out, int out_size, void* d_ws, size_t ws_size,
                              hipStream_t stream)
{
    const float* nodes  = (const float*)d_in[0];
    const float* adj    = (const float*)d_in[1];
    const float* fc_W   = (const float*)d_in[2];
    const float* fc_b   = (const float*)d_in[3];
    const float* g1_w   = (const float*)d_in[4];
    const float* g1_u   = (const float*)d_in[5];
    const float* g1_b   = (const float*)d_in[6];
    const float* g2_w   = (const float*)d_in[7];
    const float* g2_u   = (const float*)d_in[8];
    const float* g2_b   = (const float*)d_in[9];
    const float* gb_W   = (const float*)d_in[10];
    const float* gb_b   = (const float*)d_in[11];
    const float* att_W  = (const float*)d_in[12];
    const float* att_b  = (const float*)d_in[13];
    const float* att_cv = (const float*)d_in[14];
    const float* Wih    = (const float*)d_in[15];
    const float* Whh    = (const float*)d_in[16];
    const float* bih    = (const float*)d_in[17];
    const float* bhh    = (const float*)d_in[18];
    const int*   mnum   = (const int*)  d_in[19];
    float* out = (float*)d_out;

    // ---- workspace carve-up: f32 region then f16 region (~94 MB) ----
    float* h    = (float*)d_ws;              // [1024][768]
    float* h1s  = h    + HS;                 // [1024][768]
    float* agg  = h1s  + HS;                 // [3][1024][768]
    float* abuf = agg  + 3L * HS;            // [1024][512]
    float* cat  = abuf + (long)N_NOTES * ATT_D;
    float* sbuf = cat  + (long)N_NOTES * ATT_D;     // [1024][8]
    float* pooled = sbuf + (long)N_NOTES * N_HEAD;  // [64][512]
    float* xih  = pooled + (long)N_MEAS * ATT_D;    // [2][64][1024]
    float* lb0  = xih  + 2L * N_MEAS * 1024;        // [64][512]
    float* mh   = lb0  + (long)N_MEAS * ATT_D;      // [64][512]

    _Float16* adjT16 = (_Float16*)(mh + (long)N_MEAS * ATT_D); // [8][1024][1024]
    _Float16* wT16   = adjT16 + 8L * N_NOTES * N_NOTES;  // [3][768][6144]
    _Float16* uT16   = wT16   + 3L * GSIZE * KAGG;       // [3][768][768]
    _Float16* act16  = uT16   + 3L * GSIZE * GSIZE;      // [1024][6144]
    _Float16* h16    = act16  + (long)N_NOTES * KAGG;    // [1024][768]
    _Float16* hT16   = h16    + HS;                      // [768][1024]
    _Float16* rh16   = hT16   + HS;                      // [1024][768]
    _Float16* nodes16= rh16   + HS;                      // [1024][96]
    _Float16* fcW16  = nodes16 + (long)N_NOTES * IN_PAD; // [256][96]
    _Float16* gbW16  = fcW16  + (long)NOTE_H * IN_PAD;   // [768][768]
    _Float16* attW16 = gbW16  + (long)GSIZE * GSIZE;     // [512][512]
    _Float16* Wih16  = attW16 + (long)ATT_D * ATT_D;     // [2][2][1024][512]
    _Float16* cat16  = Wih16  + 4L * 1024 * ATT_D;       // [1024][512]
    _Float16* pooled16 = cat16 + (long)N_NOTES * ATT_D;  // [64][512]
    _Float16* lb016  = pooled16 + (long)N_MEAS * ATT_D;  // [64][512]

    GraphBufs G = { adjT16, wT16, uT16, act16, h16, hT16, rh16, agg };
    const int EB = (HS + 255) / 256;

    // ---- one-time layout transforms ----
    transpose16(stream, adj, (long)N_NOTES * N_NOTES, N_NOTES, N_NOTES,
                adjT16, N_NOTES, 1, (long)N_NOTES * N_NOTES, 0, N_EDGE);
    convert16(stream, nodes, N_NOTES, IN_SIZE, nodes16, IN_PAD);
    convert16(stream, fc_W, NOTE_H, IN_SIZE, fcW16, IN_PAD);
    convert16(stream, gb_W, GSIZE, GSIZE, gbW16, GSIZE);
    convert16(stream, att_W, ATT_D, ATT_D, attW16, ATT_D);
    convert16(stream, Wih, 4 * 1024, 2 * MEAS_H, Wih16, 2 * MEAS_H);

    // ---- nm_hidden = [zeros(512) | relu(nodes@fc_W^T + fc_b)] ----
    fill_zero_k<<<EB, 256, 0, stream>>>(h, HS);
    gemm16(stream, nodes16, IN_PAD, 0, fcW16, IN_PAD, 0,
           h + 2 * MEAS_H, GSIZE, 0, nullptr, 0, 0, fc_b, 0,
           N_NOTES, NOTE_H, IN_PAD, 0, 1, 1);

    // ---- graph pass 1 -> h1 ----
    graph_pass(stream, g1_w, g1_u, g1_b, h, G);
    copy_k<<<EB, 256, 0, stream>>>(h, h1s, HS);

    // ---- between = relu(h1 @ gb_W^T + gb_b) -> h ----
    convert16(stream, h, N_NOTES, GSIZE, h16, GSIZE);
    gemm16(stream, h16, GSIZE, 0, gbW16, GSIZE, 0,
           h, GSIZE, 0, nullptr, 0, 0, gb_b, 0,
           N_NOTES, GSIZE, GSIZE, 0, 1, 1);

    // ---- graph pass 2 -> h2 (stays in h) ----
    graph_pass(stream, g2_w, g2_u, g2_b, h, G);

    // ---- attention over measures ----
    build_cat_k<<<(N_NOTES * ATT_D + 255) / 256, 256, 0, stream>>>(h1s, h, cat, cat16);
    gemm16(stream, cat16, ATT_D, 0, attW16, ATT_D, 0,
           abuf, ATT_D, 0, nullptr, 0, 0, att_b, 0,
           N_NOTES, ATT_D, ATT_D, 0, 2, 1);
    att_sim_k<<<(N_NOTES * N_HEAD + 255) / 256, 256, 0, stream>>>(abuf, att_cv, sbuf);
    att_pool_k<<<N_MEAS, 256, 0, stream>>>(sbuf, cat, pooled);

    // ---- biLSTM over 64 measures (2 layers) ----
    for (int l = 0; l < 2; ++l) {
        const _Float16* xin16;
        if (l == 0) { convert16(stream, pooled, N_MEAS, ATT_D, pooled16, ATT_D); xin16 = pooled16; }
        else        { convert16(stream, lb0,    N_MEAS, ATT_D, lb016,    ATT_D); xin16 = lb016; }
        gemm16(stream, xin16, ATT_D, 0,
               Wih16 + (long)l * 2 * 1024 * (2 * MEAS_H), 2 * MEAS_H, (long)1024 * (2 * MEAS_H),
               xih, 1024, (long)N_MEAS * 1024, nullptr, 0, 0,
               bih + (long)l * 2 * 1024, 1024,
               N_MEAS, 1024, 2 * MEAS_H, 0, 0, 2);
        float* dst = (l == 0) ? lb0 : mh;
        lstm_scan_k<<<2, 256, 0, stream>>>(xih, Whh + (long)l * 2 * 1024 * MEAS_H,
                                           bhh + (long)l * 2 * 1024, dst);
    }

    // ---- assemble outputs ----
    const int TOT = N_NOTES * 1536 + N_MEAS * 512;
    write_out_k<<<(TOT + 255) / 256, 256, 0, stream>>>(h1s, h, mh, mnum, out);
    (void)in_sizes; (void)n_in; (void)out_size; (void)ws_size;
}
